// InfiniAttention_11424613007430
// MI455X (gfx1250) — compile-verified, matrix operand order checked by reading, not run on hardware
//
#include <hip/hip_runtime.h>
#include <math.h>

// ---------------- problem constants ----------------
#define S_LEN 2048
#define HID   4096
#define NH    32
#define NKV   8
#define HD    128
#define GROUPS (NH / NKV)         // 4
#define EPS   1e-8f
#define THETA 10000.0f
#define SCALE 0.08838834764831845f // 128^-0.5

typedef unsigned short u16;
typedef __attribute__((ext_vector_type(8)))  unsigned short u16x8;
typedef __attribute__((ext_vector_type(16))) unsigned short u16x16;
typedef __attribute__((ext_vector_type(16))) __bf16         v16bf;
typedef __attribute__((ext_vector_type(8)))  float          v8f;
typedef __attribute__((ext_vector_type(4)))  unsigned int   u32x4;
typedef __attribute__((ext_vector_type(8)))  int            i32x8;
typedef __attribute__((ext_vector_type(4)))  int            i32x4;

// ---------------- helpers ----------------
__device__ __forceinline__ u16 f2bf(float f) {
    unsigned u = __float_as_uint(f);
    unsigned r = (u + 0x7FFFu + ((u >> 16) & 1u)) >> 16;   // round-to-nearest-even
    return (u16)r;
}

__device__ __forceinline__ v8f wmma_bf16(u16x16 a, u16x16 b, v8f c) {
    return __builtin_amdgcn_wmma_f32_16x16x32_bf16(
        false, __builtin_bit_cast(v16bf, a),
        false, __builtin_bit_cast(v16bf, b),
        (short)0, c, false, false);
}

// Load a 16x32 bf16 fragment from a row-major GLOBAL matrix (A layout; B mirrors with N).
// Lane mapping (ISA 7.12.2, 16-bit A 16x32):
//   lane<16 : row = r0+lane,     elems 0..7 -> K=k0..k0+7,  elems 8..15 -> K=k0+16..k0+23
//   lane>=16: row = r0+lane-16,  elems 0..7 -> K=k0+8..+15, elems 8..15 -> K=k0+24..+31
__device__ __forceinline__ u16x16 ldfrag(const u16* base, int ld, int r0, int k0, int lane) {
    int row = r0 + (lane & 15);
    int kb  = k0 + ((lane >> 4) << 3);
    const u16* p = base + (size_t)row * ld + kb;
    u16x8 lo = *(const u16x8*)(p);
    u16x8 hi = *(const u16x8*)(p + 16);
    return __builtin_shufflevector(lo, hi, 0,1,2,3,4,5,6,7,8,9,10,11,12,13,14,15);
}

// Same fragment gather but from an LDS tile with padded row stride of 40 u16 (80 B):
// TDM pads 16 B per 64 B row, so bank = 20*row mod 64 -> conflict-free ds_load_b128.
#define LDS_RS 40
__device__ __forceinline__ u16x16 ldsfrag(const u16* buf, int r0, int lane) {
    int row = r0 + (lane & 15);
    int kb  = (lane >> 4) << 3;
    const u16* p = buf + row * LDS_RS + kb;
    u16x8 lo = *(const u16x8*)(p);
    u16x8 hi = *(const u16x8*)(p + 16);
    return __builtin_shufflevector(lo, hi, 0,1,2,3,4,5,6,7,8,9,10,11,12,13,14,15);
}

// ---------------- Tensor Data Mover: 2-D bf16 tile (128 rows x 32 elems) -> LDS ----------------
// D# per CDNA5 ISA:
//  group0: count=1 | lds_addr[63:32] | global_addr[120:64] | type=2 [127:126]
//  group1: data_size=1(2B) [17:16], pad_enable [20], pad_interval=3(16 DW) [24:22],
//          pad_amount=3(4 DW) [31:25], tensor_dim0 [79:48], tensor_dim1 [111:80],
//          tile_dim0=32 [127:112], tile_dim1=128 [143:128], tensor_dim0_stride [207:160]
// amdgpu-toolchain (clang-23) builtin: 6 args (g0, g1, g2, g3, g_extra, cpol).
__device__ __forceinline__ void tdm_load_tile(unsigned lds_off, const u16* gsrc,
                                              int ld_elems, int rows_total) {
    unsigned long long ga = (unsigned long long)(const void*)gsrc;
    u32x4 g0;
    g0[0] = 1u;                                        // count=1, user mode, no gather
    g0[1] = lds_off;                                   // LDS byte address
    g0[2] = (unsigned)(ga & 0xFFFFFFFFull);
    g0[3] = (unsigned)((ga >> 32) & 0x1FFFFFFull) | (2u << 30);  // addr[56:32] | type=2
    i32x8 g1;
    g1[0] = (1 << 16) | (1 << 20) | (3 << 22) | (3 << 25);       // 2B, pad: every 16DW add 4DW
    g1[1] = (int)(((unsigned)ld_elems & 0xFFFFu) << 16);         // tensor_dim0[15:0]
    g1[2] = (int)((((unsigned)ld_elems >> 16) & 0xFFFFu) |
                  (((unsigned)rows_total & 0xFFFFu) << 16));     // dim0 hi | dim1 lo
    g1[3] = (int)((((unsigned)rows_total >> 16) & 0xFFFFu) |
                  (32u << 16));                                  // dim1 hi | tile_dim0=32
    g1[4] = 128;                                                 // tile_dim1=128, tile_dim2=0
    g1[5] = ld_elems;                                            // tensor_dim0_stride lo
    g1[6] = 0;
    g1[7] = 0;
    i32x4 z4 = {0, 0, 0, 0};
    i32x8 z8 = {0, 0, 0, 0, 0, 0, 0, 0};
    __builtin_amdgcn_tensor_load_to_lds(g0, g1, z4, z4, z8, 0);
}

// ---------------- elementwise converts ----------------
__global__ void cvt_f32_to_bf16(const float* __restrict__ src, u16* __restrict__ dst, int n) {
    int i = blockIdx.x * blockDim.x + threadIdx.x;
    if (i < n) dst[i] = f2bf(src[i]);
}

// W [K][N] fp32 row-major  ->  Wt [N][K] bf16 bits
__global__ void wtrans_bf16(const float* __restrict__ W, u16* __restrict__ Wt, int K, int N) {
    __shared__ float tile[32][33];
    int n0 = blockIdx.x * 32, k0 = blockIdx.y * 32;
    int tx = threadIdx.x, ty = threadIdx.y;       // 32 x 8
    #pragma unroll
    for (int i = 0; i < 32; i += 8)
        tile[ty + i][tx] = W[(size_t)(k0 + ty + i) * N + n0 + tx];
    __syncthreads();
    #pragma unroll
    for (int i = 0; i < 32; i += 8)
        Wt[(size_t)(n0 + ty + i) * K + k0 + tx] = f2bf(tile[tx][ty + i]);
}

// ---------------- WMMA GEMM with TDM-staged LDS double buffering ----------------
// C[M][N] f32 = A[M][K] bf16 * Bt[N][K] bf16.  Block tile 128x128, BK=32.
// 8 waves as 4x2; each wave owns 32x64 (2x4 wmma accumulators).
// Wave 0 drives the Tensor Data Mover; TENSORcnt + one s_barrier per K-step.
__global__ __launch_bounds__(256) void gemm_bf16_tdm(const u16* __restrict__ A,
                                                     const u16* __restrict__ Bt,
                                                     float* __restrict__ C,
                                                     int M, int N, int K) {
    __shared__ u16 shA[2][128 * LDS_RS];
    __shared__ u16 shB[2][128 * LDS_RS];
    int lane = threadIdx.x & 31;
    int wave = threadIdx.x >> 5;
    int bm0 = blockIdx.y * 128;
    int bn0 = blockIdx.x * 128;
    int wm = (wave >> 1) * 32;       // 0,32,64,96
    int wn = (wave & 1) * 64;        // 0,64

    v8f acc[2][4] = {};
    int nk = K / 32;

    if (wave == 0) {
        tdm_load_tile((unsigned)(unsigned long long)(const void*)shA[0],
                      A + (size_t)bm0 * K, K, M);
        tdm_load_tile((unsigned)(unsigned long long)(const void*)shB[0],
                      Bt + (size_t)bn0 * K, K, N);
    }
    for (int kk = 0; kk < nk; ++kk) {
        int cur = kk & 1;
        if (wave == 0) __builtin_amdgcn_s_wait_tensorcnt(0);  // buf[cur] resident
        __syncthreads();   // publish buf[cur]; also: everyone done reading buf[cur^1]
        if (wave == 0 && kk + 1 < nk) {
            int k1 = (kk + 1) * 32;
            tdm_load_tile((unsigned)(unsigned long long)(const void*)shA[cur ^ 1],
                          A + (size_t)bm0 * K + k1, K, M);
            tdm_load_tile((unsigned)(unsigned long long)(const void*)shB[cur ^ 1],
                          Bt + (size_t)bn0 * K + k1, K, N);
        }
        u16x16 a0 = ldsfrag(shA[cur], wm, lane);
        u16x16 a1 = ldsfrag(shA[cur], wm + 16, lane);
        u16x16 b0 = ldsfrag(shB[cur], wn, lane);
        u16x16 b1 = ldsfrag(shB[cur], wn + 16, lane);
        u16x16 b2 = ldsfrag(shB[cur], wn + 32, lane);
        u16x16 b3 = ldsfrag(shB[cur], wn + 48, lane);
        acc[0][0] = wmma_bf16(a0, b0, acc[0][0]);
        acc[0][1] = wmma_bf16(a0, b1, acc[0][1]);
        acc[0][2] = wmma_bf16(a0, b2, acc[0][2]);
        acc[0][3] = wmma_bf16(a0, b3, acc[0][3]);
        acc[1][0] = wmma_bf16(a1, b0, acc[1][0]);
        acc[1][1] = wmma_bf16(a1, b1, acc[1][1]);
        acc[1][2] = wmma_bf16(a1, b2, acc[1][2]);
        acc[1][3] = wmma_bf16(a1, b3, acc[1][3]);
    }
    int col = lane & 15, rofs = (lane >> 4) * 8;
    #pragma unroll
    for (int i = 0; i < 2; ++i)
        #pragma unroll
        for (int j = 0; j < 4; ++j)
            #pragma unroll
            for (int r = 0; r < 8; ++r)
                C[(size_t)(bm0 + wm + i * 16 + rofs + r) * N + bn0 + wn + j * 16 + col]
                    = acc[i][j][r];
}

// ---------------- memory retrieval: comb = gate * (qn@M)/(qn@N^T+eps) ----------------
__global__ __launch_bounds__(256) void mem_retrieve(const float* __restrict__ Qf,
                                                    const float* __restrict__ memory,
                                                    const float* __restrict__ norm_m,
                                                    const float* __restrict__ beta,
                                                    float* __restrict__ comb) {
    __shared__ float qn[64 * 128];
    __shared__ float rsum[64];
    int h  = blockIdx.x;              // 0..31
    int s0 = blockIdx.y * 64;
    int t  = threadIdx.x;
    const float* mem = memory + (size_t)(h & (NKV - 1)) * HD * HD;  // tile: h % NKV
    const float* nm  = norm_m + (size_t)(h & (NKV - 1)) * HD * HD;
    for (int it = 0; it < 32; ++it) {
        int lin = it * 256 + t, row = lin >> 7, d = lin & 127;
        float q = Qf[(size_t)(s0 + row) * (NH * HD) + h * HD + d];
        float f = (q > 0.f ? q : (__expf(q) - 1.f)) + 1.f;          // elu + 1
        qn[row * 128 + d] = f;
    }
    __syncthreads();
    if (t < 64) {
        float s = 0.f;
        for (int d = 0; d < 128; ++d) s += qn[t * 128 + d];
        rsum[t] = 1.f / (s + EPS);
    }
    __syncthreads();
    for (int it = 0; it < 32; ++it) {
        int lin = it * 256 + t, row = lin >> 7, d = lin & 127;
        qn[row * 128 + d] *= rsum[row];
    }
    __syncthreads();
    float gate = 1.f / (1.f + __expf(-beta[0]));
    for (int it = 0; it < 32; ++it) {
        int lin = it * 256 + t, row = lin >> 7, e = lin & 127;
        float num = 0.f, den = 0.f;
        for (int d = 0; d < 128; ++d) {
            float q = qn[row * 128 + d];
            num += q * mem[d * 128 + e];
            den += q * nm[e * 128 + d];                             // norm_t[d][e] = norm_m[e][d]
        }
        comb[(size_t)(s0 + row) * (NH * HD) + h * HD + e] = gate * (num / (den + EPS));
    }
}

// ---------------- kn / dv pass for delta rule ----------------
__global__ __launch_bounds__(256) void kn_dv(const float* __restrict__ Kf,
                                             const float* __restrict__ Vf,
                                             const float* __restrict__ memory,
                                             const float* __restrict__ norm_m,
                                             float* __restrict__ knbuf,
                                             float* __restrict__ dvbuf,
                                             float* __restrict__ colsum) {
    __shared__ float kn[64 * 128];
    __shared__ float rsum[64];
    int h  = blockIdx.x;              // 0..7
    int s0 = blockIdx.y * 64;
    int t  = threadIdx.x;
    const float* mem = memory + (size_t)h * HD * HD;
    const float* nm  = norm_m + (size_t)h * HD * HD;
    for (int it = 0; it < 32; ++it) {
        int lin = it * 256 + t, row = lin >> 7, d = lin & 127;
        float k = Kf[(size_t)(s0 + row) * (NKV * HD) + h * HD + d];
        float f = (k > 0.f ? k : (__expf(k) - 1.f)) + 1.f;
        kn[row * 128 + d] = f;
    }
    __syncthreads();
    if (t < 64) {
        float s = 0.f;
        for (int d = 0; d < 128; ++d) s += kn[t * 128 + d];
        rsum[t] = 1.f / (s + EPS);
    }
    __syncthreads();
    for (int it = 0; it < 32; ++it) {
        int lin = it * 256 + t, row = lin >> 7, d = lin & 127;
        float v = kn[row * 128 + d] * rsum[row];
        kn[row * 128 + d] = v;
        knbuf[(size_t)h * S_LEN * HD + (size_t)(s0 + row) * HD + d] = v;
        atomicAdd(&colsum[h * HD + d], v);
    }
    __syncthreads();
    for (int it = 0; it < 32; ++it) {
        int lin = it * 256 + t, row = lin >> 7, e = lin & 127;
        float num = 0.f, den = 0.f;
        for (int d = 0; d < 128; ++d) {
            float q = kn[row * 128 + d];
            num += q * mem[d * 128 + e];
            den += q * nm[e * 128 + d];
        }
        float vret = num / (den + EPS);
        float v = Vf[(size_t)(s0 + row) * (NKV * HD) + h * HD + e];
        dvbuf[(size_t)h * S_LEN * HD + (size_t)(s0 + row) * HD + e] = v - vret;
    }
}

__global__ void zero_buf(float* __restrict__ p, int n) {
    int i = blockIdx.x * blockDim.x + threadIdx.x;
    if (i < n) p[i] = 0.f;
}

// ---------------- delta = kn^T @ dv, write updated memory / norm ----------------
__global__ __launch_bounds__(256) void delta_update(const float* __restrict__ knbuf,
                                                    const float* __restrict__ dvbuf,
                                                    const float* __restrict__ memory,
                                                    const float* __restrict__ norm_m,
                                                    const float* __restrict__ colsum,
                                                    float* __restrict__ out_mem,
                                                    float* __restrict__ out_norm) {
    __shared__ float sk[16][17];
    __shared__ float sd[16][17];
    int h  = blockIdx.z;
    int dt = blockIdx.y * 16;
    int et = blockIdx.x * 16;
    int tx = threadIdx.x & 15, ty = threadIdx.x >> 4;
    const float* kh = knbuf + (size_t)h * S_LEN * HD;
    const float* dh = dvbuf + (size_t)h * S_LEN * HD;
    float acc = 0.f;
    for (int s = 0; s < S_LEN; s += 16) {
        sk[ty][tx] = kh[(size_t)(s + ty) * HD + dt + tx];
        sd[ty][tx] = dh[(size_t)(s + ty) * HD + et + tx];
        __syncthreads();
        #pragma unroll
        for (int j = 0; j < 16; ++j) acc += sk[j][ty] * sd[j][tx];
        __syncthreads();
    }
    int d = dt + ty, e = et + tx;
    size_t idx = (size_t)h * HD * HD + (size_t)d * HD + e;
    out_mem[idx]  = memory[idx] + acc;
    out_norm[idx] = norm_m[idx] + colsum[h * HD + e];
}

// ---------------- RoPE + layout converts ----------------
__global__ void rope_pack(const float* __restrict__ X, const int* __restrict__ pid,
                          u16* __restrict__ out, int nheads) {
    int idx = blockIdx.x * blockDim.x + threadIdx.x;
    int total = S_LEN * nheads * (HD / 2);
    if (idx >= total) return;
    int j = idx & 63;
    int h = (idx >> 6) % nheads;
    int s = idx / (64 * nheads);
    float pos = (float)pid[s];
    float inv = __powf(THETA, -(float)(2 * j) / (float)HD);
    float ang = pos * inv, c = __cosf(ang), sn = __sinf(ang);
    size_t base = (size_t)s * nheads * HD + (size_t)h * HD;
    float x0 = X[base + j], x1 = X[base + j + 64];
    size_t ob = (size_t)h * S_LEN * HD + (size_t)s * HD;
    out[ob + j]      = f2bf(x0 * c - x1 * sn);
    out[ob + j + 64] = f2bf(x1 * c + x0 * sn);
}

__global__ void v_transpose_bf16(const float* __restrict__ Vf, u16* __restrict__ Vt) {
    int idx = blockIdx.x * blockDim.x + threadIdx.x;
    if (idx >= NKV * S_LEN * HD) return;
    int d = idx & 127;
    int s = (idx >> 7) & (S_LEN - 1);
    int h = idx >> 18;
    Vt[(size_t)h * HD * S_LEN + (size_t)d * S_LEN + s] =
        f2bf(Vf[(size_t)s * (NKV * HD) + h * HD + d]);
}

// ---------------- flash causal attention (WMMA), adds (1-gate)*O into comb ----------------
__global__ __launch_bounds__(256) void flash_attn(const u16* __restrict__ Qr,
                                                  const u16* __restrict__ Kr,
                                                  const u16* __restrict__ Vt,
                                                  const float* __restrict__ beta,
                                                  float* __restrict__ comb) {
    __shared__ u16 pbuf[8][16 * 32];         // per-wave P staging (row-major 16x32)
    int lane = threadIdx.x & 31;
    int wave = threadIdx.x >> 5;
    int g    = blockIdx.x * 8 + wave;
    int head = g >> 7;                       // S/16 = 128 query tiles per head
    int qt   = g & 127;
    int hkv  = head / GROUPS;                // repeat_interleave GQA
    const u16* Qh = Qr + (size_t)head * S_LEN * HD;
    const u16* Kh = Kr + (size_t)hkv  * S_LEN * HD;
    const u16* Vh = Vt + (size_t)hkv  * HD * S_LEN;

    u16x16 qf[4];
    #pragma unroll
    for (int c = 0; c < 4; ++c) qf[c] = ldfrag(Qh, HD, qt * 16, c * 32, lane);

    v8f o[8] = {};
    float m[8], l[8];
    #pragma unroll
    for (int r = 0; r < 8; ++r) { m[r] = -3.0e38f; l[r] = 0.f; }

    int col = lane & 15, half = lane >> 4, rofs = half * 8;
    int nkb = (qt * 16 + 15) / 32 + 1;       // causal: only key blocks <= last query

    for (int kb = 0; kb < nkb; ++kb) {
        int k0 = kb * 32;
        v8f s0 = {}, s1 = {};
        #pragma unroll
        for (int c = 0; c < 4; ++c) {
            u16x16 b0 = ldfrag(Kh, HD, k0,      c * 32, lane);
            u16x16 b1 = ldfrag(Kh, HD, k0 + 16, c * 32, lane);
            s0 = wmma_bf16(qf[c], b0, s0);
            s1 = wmma_bf16(qf[c], b1, s1);
        }
        float mm[8];
        #pragma unroll
        for (int r = 0; r < 8; ++r) {
            int qrow = qt * 16 + rofs + r;
            float v0 = (k0 + col      <= qrow) ? s0[r] * SCALE : -3.0e38f;
            float v1 = (k0 + 16 + col <= qrow) ? s1[r] * SCALE : -3.0e38f;
            s0[r] = v0; s1[r] = v1;
            mm[r] = fmaxf(v0, v1);
        }
        #pragma unroll
        for (int off = 1; off < 16; off <<= 1)
            #pragma unroll
            for (int r = 0; r < 8; ++r)
                mm[r] = fmaxf(mm[r], __shfl_xor(mm[r], off, 32));
        float alpha[8], ls[8];
        #pragma unroll
        for (int r = 0; r < 8; ++r) {
            float mn = fmaxf(m[r], mm[r]);
            alpha[r] = __expf(m[r] - mn);
            m[r] = mn;
            float p0 = __expf(s0[r] - mn), p1 = __expf(s1[r] - mn);
            s0[r] = p0; s1[r] = p1;
            ls[r] = p0 + p1;
        }
        #pragma unroll
        for (int off = 1; off < 16; off <<= 1)
            #pragma unroll
            for (int r = 0; r < 8; ++r)
                ls[r] += __shfl_xor(ls[r], off, 32);
        #pragma unroll
        for (int r = 0; r < 8; ++r) l[r] = l[r] * alpha[r] + ls[r];
        #pragma unroll
        for (int t = 0; t < 8; ++t)
            #pragma unroll
            for (int r = 0; r < 8; ++r) o[t][r] *= alpha[r];

        // stage P (D-layout -> row-major LDS), per-wave, no block barrier needed
        u16* pb = pbuf[wave];
        #pragma unroll
        for (int r = 0; r < 8; ++r) {
            int row = rofs + r;
            pb[row * 32 + col]      = f2bf(s0[r]);
            pb[row * 32 + 16 + col] = f2bf(s1[r]);
        }
        asm volatile("s_wait_dscnt 0x0" ::: "memory");
        u16x16 pf;
        {
            int row = lane & 15, kbse = half * 8;
            const u16* pr = pb + row * 32;
            #pragma unroll
            for (int i = 0; i < 8; ++i) pf[i] = pr[kbse + i];
            #pragma unroll
            for (int i = 0; i < 8; ++i) pf[8 + i] = pr[16 + kbse + i];
        }
        asm volatile("s_wait_dscnt 0x0" ::: "memory");
        #pragma unroll
        for (int nt = 0; nt < 8; ++nt) {
            u16x16 bv = ldfrag(Vh, S_LEN, nt * 16, k0, lane); // "rows"=d, "K"=keys
            o[nt] = wmma_bf16(pf, bv, o[nt]);
        }
    }

    float g1 = 1.f - 1.f / (1.f + __expf(-beta[0]));
    #pragma unroll
    for (int nt = 0; nt < 8; ++nt)
        #pragma unroll
        for (int r = 0; r < 8; ++r) {
            int srow = qt * 16 + rofs + r;
            int d    = nt * 16 + col;
            size_t idx = (size_t)srow * (NH * HD) + (size_t)head * HD + d;
            comb[idx] += g1 * (o[nt][r] / l[r]);
        }
}

// ---------------- workspace layout (bytes) ----------------
#define OF_HID   ((size_t)0)                                   // bf16 hidden (reused for comb bf16)
#define OF_WQT   (OF_HID + (size_t)S_LEN * HID * 2)
#define OF_WKT   (OF_WQT + (size_t)HID * (NH * HD) * 2)
#define OF_WVT   (OF_WKT + (size_t)HID * (NKV * HD) * 2)
#define OF_WOT   (OF_WVT + (size_t)HID * (NKV * HD) * 2)
#define OF_Q     (OF_WOT + (size_t)(NH * HD) * HID * 2)
#define OF_K     (OF_Q   + (size_t)S_LEN * (NH * HD) * 4)
#define OF_V     (OF_K   + (size_t)S_LEN * (NKV * HD) * 4)
#define OF_QR    (OF_V   + (size_t)S_LEN * (NKV * HD) * 4)
#define OF_KR    (OF_QR  + (size_t)NH * S_LEN * HD * 2)
#define OF_VT    (OF_KR  + (size_t)NKV * S_LEN * HD * 2)
#define OF_COMB  (OF_VT  + (size_t)NKV * S_LEN * HD * 2)
#define OF_KN    (OF_COMB + (size_t)S_LEN * (NH * HD) * 4)
#define OF_DV    (OF_KN  + (size_t)NKV * S_LEN * HD * 4)
#define OF_CS    (OF_DV  + (size_t)NKV * S_LEN * HD * 4)

extern "C" void kernel_launch(void* const* d_in, const int* in_sizes, int n_in,
                              void* d_out, int out_size, void* d_ws, size_t ws_size,
                              hipStream_t stream) {
    (void)in_sizes; (void)n_in; (void)out_size; (void)ws_size;
    const float* hidden = (const float*)d_in[0];
    const int*   pid    = (const int*)  d_in[1];
    const float* wq     = (const float*)d_in[2];
    const float* wk     = (const float*)d_in[3];
    const float* wv     = (const float*)d_in[4];
    const float* wo     = (const float*)d_in[5];
    const float* memory = (const float*)d_in[6];
    const float* norm_m = (const float*)d_in[7];
    const float* beta   = (const float*)d_in[8];

    char* ws = (char*)d_ws;
    u16*   hid_bf  = (u16*)  (ws + OF_HID);
    u16*   wqT     = (u16*)  (ws + OF_WQT);
    u16*   wkT     = (u16*)  (ws + OF_WKT);
    u16*   wvT     = (u16*)  (ws + OF_WVT);
    u16*   woT     = (u16*)  (ws + OF_WOT);
    float* Qf      = (float*)(ws + OF_Q);
    float* Kf      = (float*)(ws + OF_K);
    float* Vf      = (float*)(ws + OF_V);
    u16*   Qr      = (u16*)  (ws + OF_QR);
    u16*   Kr      = (u16*)  (ws + OF_KR);
    u16*   Vt      = (u16*)  (ws + OF_VT);
    float* comb    = (float*)(ws + OF_COMB);
    float* knbuf   = (float*)(ws + OF_KN);
    float* dvbuf   = (float*)(ws + OF_DV);
    float* colsum  = (float*)(ws + OF_CS);
    u16*   comb_bf = hid_bf;                    // reuse: hidden bf16 dead after QKV GEMMs

    float* out      = (float*)d_out;                       // [S][HID]
    float* out_mem  = out + (size_t)S_LEN * HID;           // [NKV][HD][HD]
    float* out_norm = out_mem + (size_t)NKV * HD * HD;     // [NKV][HD][HD]

    // 1) converts / transposes
    {
        int n = S_LEN * HID;
        cvt_f32_to_bf16<<<(n + 255) / 256, 256, 0, stream>>>(hidden, hid_bf, n);
        dim3 tb(32, 8);
        wtrans_bf16<<<dim3((NH * HD) / 32, HID / 32), tb, 0, stream>>>(wq, wqT, HID, NH * HD);
        wtrans_bf16<<<dim3((NKV * HD) / 32, HID / 32), tb, 0, stream>>>(wk, wkT, HID, NKV * HD);
        wtrans_bf16<<<dim3((NKV * HD) / 32, HID / 32), tb, 0, stream>>>(wv, wvT, HID, NKV * HD);
        wtrans_bf16<<<dim3(HID / 32, (NH * HD) / 32), tb, 0, stream>>>(wo, woT, NH * HD, HID);
    }
    // 2) QKV projections (TDM-staged WMMA GEMM)
    gemm_bf16_tdm<<<dim3((NH * HD) / 128,  S_LEN / 128), 256, 0, stream>>>(hid_bf, wqT, Qf,
                                                                           S_LEN, NH * HD,  HID);
    gemm_bf16_tdm<<<dim3((NKV * HD) / 128, S_LEN / 128), 256, 0, stream>>>(hid_bf, wkT, Kf,
                                                                           S_LEN, NKV * HD, HID);
    gemm_bf16_tdm<<<dim3((NKV * HD) / 128, S_LEN / 128), 256, 0, stream>>>(hid_bf, wvT, Vf,
                                                                           S_LEN, NKV * HD, HID);

    // 3) compressive-memory retrieval -> comb = gate * mem_out   (pre-RoPE Q)
    mem_retrieve<<<dim3(NH, S_LEN / 64), 256, 0, stream>>>(Qf, memory, norm_m, beta, comb);

    // 4) delta-rule update -> out_mem / out_norm
    zero_buf<<<(NKV * HD + 255) / 256, 256, 0, stream>>>(colsum, NKV * HD);
    kn_dv<<<dim3(NKV, S_LEN / 64), 256, 0, stream>>>(Kf, Vf, memory, norm_m, knbuf, dvbuf, colsum);
    delta_update<<<dim3(HD / 16, HD / 16, NKV), 256, 0, stream>>>(knbuf, dvbuf, memory, norm_m,
                                                                  colsum, out_mem, out_norm);

    // 5) RoPE + layout for attention
    {
        int nq = S_LEN * NH * (HD / 2);
        rope_pack<<<(nq + 255) / 256, 256, 0, stream>>>(Qf, pid, Qr, NH);
        int nk = S_LEN * NKV * (HD / 2);
        rope_pack<<<(nk + 255) / 256, 256, 0, stream>>>(Kf, pid, Kr, NKV);
        int nv = NKV * S_LEN * HD;
        v_transpose_bf16<<<(nv + 255) / 256, 256, 0, stream>>>(Vf, Vt);
    }

    // 6) flash causal GQA attention, accumulates (1-gate)*attn into comb
    flash_attn<<<(NH * (S_LEN / 16)) / 8, 256, 0, stream>>>(Qr, Kr, Vt, beta, comb);

    // 7) output projection: out = comb(bf16) @ wo
    {
        int n = S_LEN * HID;
        cvt_f32_to_bf16<<<(n + 255) / 256, 256, 0, stream>>>(comb, comb_bf, n);
        gemm_bf16_tdm<<<dim3(HID / 128, S_LEN / 128), 256, 0, stream>>>(comb_bf, woT, out,
                                                                        S_LEN, HID, NH * HD);
    }
}
